// HGNN_37546604102398
// MI455X (gfx1250) — compile-verified, matrix operand order checked by reading, not run on hardware
//
#include <hip/hip_runtime.h>
#include <hip/hip_bf16.h>
#include <math.h>

typedef __attribute__((ext_vector_type(16))) __bf16 v16bf;
typedef __attribute__((ext_vector_type(8)))  float  v8f;

#define NNODES 50000
#define NEDGES 150000
#define FIN    512
#define HD     128
#define NH     2
#define NG     64

// A/B fragment K-index map for 16-bit WMMA (per ISA 7.12.2):
// lane<16 -> base 0, lane>=16 -> base 8 ; elems 0..7 -> K=base+i ; elems 8..15 -> K=16+base+(i-8)
__device__ __forceinline__ int kmap(int lane, int i) {
    int base = (lane >= 16) ? 8 : 0;
    return (i < 8) ? (base + i) : (16 + base + (i - 8));
}

// ---------------- weight packing ----------------
// Pack row-major KxN fp32 weight into B-fragment-ordered bf16:
// dst[((nt*Kt + kt)*32 + lane)*16 + i]
__global__ void pack_b(const float* __restrict__ src, __bf16* __restrict__ dst, int K, int N) {
    int lane = threadIdx.x;
    int kt = blockIdx.x, nt = blockIdx.y, Kt = gridDim.x;
    int n = nt * 16 + (lane & 15);
    __bf16* o = dst + (((size_t)nt * Kt + kt) * 32 + lane) * 16;
    for (int i = 0; i < 16; i++) {
        int k = kt * 32 + kmap(lane, i);
        float v = (k < K && n < N) ? src[(size_t)k * N + n] : 0.f;
        o[i] = (__bf16)v;
    }
}

// Pack per-relation (H=2, 64, 64) head transform as 128x128 block-diagonal weight.
__global__ void pack_rel(const float* __restrict__ rel, __bf16* __restrict__ dst) {
    int lane = threadIdx.x;
    int kt = blockIdx.x, nt = blockIdx.y; // Kt=4, Nt=8
    int n = nt * 16 + (lane & 15);
    __bf16* o = dst + (((size_t)nt * 4 + kt) * 32 + lane) * 16;
    for (int i = 0; i < 16; i++) {
        int k = kt * 32 + kmap(lane, i);
        float v = ((k >> 6) == (n >> 6)) ? rel[(((k >> 6) * 64) + (k & 63)) * 64 + (n & 63)] : 0.f;
        o[i] = (__bf16)v;
    }
}

__global__ void f32_to_bf16(const float* __restrict__ s, __bf16* __restrict__ d, long n) {
    long i = (long)blockIdx.x * blockDim.x + threadIdx.x;
    if (i < n) d[i] = (__bf16)s[i];
}

// ---------------- WMMA GEMM ----------------
// One wave computes a 16x64 C tile, software-pipelined (double-buffered fragments)
// so next K-step's global loads overlap with the current WMMAs.
// A: bf16 row-major (stride sA). B: packed bf16.
// mode 0: +bias ; mode 2: +bias, relu ; mode 1: +bias, sigmoid(skip)-gated residual, relu.
__global__ void gemm_wmma(const __bf16* __restrict__ A, int sA,
                          const __bf16* __restrict__ B, int K,
                          const float* __restrict__ bias,
                          float* __restrict__ Cf, __bf16* __restrict__ Cb, int sC,
                          int M,
                          const float* __restrict__ resid, int sR,
                          const float* __restrict__ skipPtr, int skipIdx, int mode) {
    int lane = threadIdx.x;
    int mt = blockIdx.x;
    int nt4 = blockIdx.y;
    int row = mt * 16 + (lane & 15);
    int rowc = row < M ? row : (M - 1);
    int baseK = (lane >> 4) * 8;
    int Kt = K >> 5;
    const __bf16* arow = A + (size_t)rowc * sA + baseK;
    // per-lane base for packed B: elem idx = (nt4*4+t)*Kt*512 + kt*512 + lane*16
    const __bf16* pB = B + (size_t)(nt4 * 4) * (size_t)Kt * 512 + (size_t)lane * 16;
    const size_t bstep = (size_t)Kt * 512;

    v8f acc0 = {}, acc1 = {}, acc2 = {}, acc3 = {};

    auto loadA = [&](int kt, v16bf& a) {
        const __bf16* p0 = arow + kt * 32;
#pragma unroll
        for (int i = 0; i < 8; i++) { a[i] = p0[i]; a[8 + i] = p0[16 + i]; }
    };
    auto loadB = [&](int kt, v16bf& b0, v16bf& b1, v16bf& b2, v16bf& b3) {
        size_t o = (size_t)kt * 512;
        b0 = *(const v16bf*)(pB + o);
        b1 = *(const v16bf*)(pB + bstep + o);
        b2 = *(const v16bf*)(pB + 2 * bstep + o);
        b3 = *(const v16bf*)(pB + 3 * bstep + o);
    };

    v16bf a0, b00, b01, b02, b03;
    loadA(0, a0);
    loadB(0, b00, b01, b02, b03);
    for (int kt = 0; kt < Kt; ++kt) {
        v16bf a1, b10, b11, b12, b13;
        int kn = (kt + 1 < Kt) ? (kt + 1) : kt; // last iter: harmless reload
        loadA(kn, a1);
        loadB(kn, b10, b11, b12, b13);
        acc0 = __builtin_amdgcn_wmma_f32_16x16x32_bf16(false, a0, false, b00, (short)0, acc0, false, false);
        acc1 = __builtin_amdgcn_wmma_f32_16x16x32_bf16(false, a0, false, b01, (short)0, acc1, false, false);
        acc2 = __builtin_amdgcn_wmma_f32_16x16x32_bf16(false, a0, false, b02, (short)0, acc2, false, false);
        acc3 = __builtin_amdgcn_wmma_f32_16x16x32_bf16(false, a0, false, b03, (short)0, acc3, false, false);
        a0 = a1; b00 = b10; b01 = b11; b02 = b12; b03 = b13;
    }

    float gate = 0.f;
    if (mode == 1) gate = 1.f / (1.f + __expf(-skipPtr[skipIdx]));
    int moff = (lane >> 4) * 8;
#pragma unroll
    for (int t = 0; t < 4; t++) {
        const v8f& acc = (t == 0) ? acc0 : (t == 1) ? acc1 : (t == 2) ? acc2 : acc3;
        int n = nt4 * 64 + t * 16 + (lane & 15);
        float bv = bias ? bias[n] : 0.f;
#pragma unroll
        for (int j = 0; j < 8; j++) {
            int m = mt * 16 + moff + j;
            if (m < M) {
                float v = acc[j] + bv;
                if (mode == 1) v = gate * v + (1.f - gate) * resid[(size_t)m * sR + n];
                if (mode == 1 || mode == 2) v = fmaxf(v, 0.f);
                if (Cf) Cf[(size_t)m * sC + n] = v;
                if (Cb) Cb[(size_t)m * sC + n] = (__bf16)v;
            }
        }
    }
}

// ---------------- edge phase ----------------
__device__ __forceinline__ unsigned ford(float f) {
    unsigned u = __float_as_uint(f);
    return (u & 0x80000000u) ? ~u : (u | 0x80000000u);
}
__device__ __forceinline__ float funord(unsigned u) {
    return __uint_as_float((u & 0x80000000u) ? (u & 0x7fffffffu) : ~u);
}

// idx = ((r*E + e)*2 + h)
__global__ void edge_logits(const int* __restrict__ e0, const int* __restrict__ e1,
                            const float* __restrict__ Qf,
                            const float* __restrict__ KR0, const float* __restrict__ KR1,
                            const float* __restrict__ prel,
                            float* __restrict__ logits, unsigned* __restrict__ maxb) {
    int idx = blockIdx.x * blockDim.x + threadIdx.x;
    if (idx >= 2 * NEDGES * NH) return;
    int h = idx & 1;
    int e = (idx >> 1) % NEDGES;
    int r = (idx >> 1) / NEDGES;
    const int* ep = r ? e1 : e0;
    int src = ep[e];
    int dst = ep[NEDGES + e];
    const float4* q4 = (const float4*)(Qf + (size_t)dst * HD + h * 64);
    const float4* k4 = (const float4*)((r ? KR1 : KR0) + (size_t)src * HD + h * 64);
    float s = 0.f;
#pragma unroll
    for (int d = 0; d < 16; d++) {
        float4 a = q4[d], b = k4[d];
        s += a.x * b.x + a.y * b.y + a.z * b.z + a.w * b.w;
    }
    s *= prel[r * NH + h] * 0.125f; // 1/sqrt(64)
    logits[idx] = s;
    atomicMax(&maxb[((size_t)r * NNODES + dst) * NH + h], ford(s));
}

__global__ void edge_accum(const int* __restrict__ e0, const int* __restrict__ e1,
                           const float* __restrict__ VR0, const float* __restrict__ VR1,
                           const float* __restrict__ logits, const unsigned* __restrict__ maxb,
                           float* __restrict__ sumb,
                           float* __restrict__ num0, float* __restrict__ num1) {
    int idx = blockIdx.x * blockDim.x + threadIdx.x;
    if (idx >= 2 * NEDGES * NH) return;
    int h = idx & 1;
    int e = (idx >> 1) % NEDGES;
    int r = (idx >> 1) / NEDGES;
    const int* ep = r ? e1 : e0;
    int src = ep[e];
    int dst = ep[NEDGES + e];
    float m = funord(maxb[((size_t)r * NNODES + dst) * NH + h]);
    float ev = __expf(logits[idx] - m);
    atomicAdd(&sumb[((size_t)r * NNODES + dst) * NH + h], ev);
    const float4* v4 = (const float4*)((r ? VR1 : VR0) + (size_t)src * HD + h * 64);
    float* nb = (r ? num1 : num0) + (size_t)dst * HD + h * 64;
#pragma unroll
    for (int d = 0; d < 16; d++) {
        float4 t = v4[d];
        atomicAdd(&nb[4 * d + 0], ev * t.x);
        atomicAdd(&nb[4 * d + 1], ev * t.y);
        atomicAdd(&nb[4 * d + 2], ev * t.z);
        atomicAdd(&nb[4 * d + 3], ev * t.w);
    }
}

// normalize per relation, sum, exact GELU, emit bf16
__global__ void node_gelu(const float* __restrict__ num0, const float* __restrict__ num1,
                          const float* __restrict__ sumb, __bf16* __restrict__ aggBf) {
    long idx = (long)blockIdx.x * blockDim.x + threadIdx.x;
    if (idx >= (long)NNODES * HD) return;
    int n = (int)(idx / HD);
    int h = ((int)(idx % HD)) >> 6;
    float s0 = sumb[((size_t)0 * NNODES + n) * NH + h];
    float s1 = sumb[((size_t)1 * NNODES + n) * NH + h];
    float v = num0[idx] / (s0 + 1e-16f) + num1[idx] / (s1 + 1e-16f);
    v = 0.5f * v * (1.f + erff(v * 0.70710678118654752f));
    aggBf[idx] = (__bf16)v;
}

// ---------------- pooling + MLP ----------------
__global__ void pool_sum(const float* __restrict__ Hf, const int* __restrict__ batch,
                         float* __restrict__ ps, float* __restrict__ cnt) {
    long idx = (long)blockIdx.x * blockDim.x + threadIdx.x;
    if (idx >= (long)NNODES * HD) return;
    int n = (int)(idx / HD), c = (int)(idx % HD);
    int g = batch[n];
    atomicAdd(&ps[(size_t)g * HD + c], Hf[idx]);
    if (c == 0) atomicAdd(&cnt[g], 1.f);
}

__global__ void mlp1(const float* __restrict__ ps, const float* __restrict__ cnt,
                     const float* __restrict__ W, const float* __restrict__ b,
                     float* __restrict__ hid) {
    int idx = blockIdx.x * blockDim.x + threadIdx.x;
    if (idx >= NG * HD) return;
    int g = idx / HD, o = idx % HD;
    float c = fmaxf(cnt[g], 1.f);
    float s = b[o];
    for (int j = 0; j < HD; j++) s += (ps[(size_t)g * HD + j] / c) * W[(size_t)j * HD + o];
    hid[idx] = fmaxf(s, 0.f);
}

__global__ void mlp2(const float* __restrict__ hid, const float* __restrict__ W,
                     const float* __restrict__ b, float* __restrict__ out) {
    int idx = blockIdx.x * blockDim.x + threadIdx.x;
    if (idx >= NG * 256) return;
    int g = idx / 256, o = idx % 256;
    float s = b[o];
    for (int j = 0; j < HD; j++) s += hid[(size_t)g * HD + j] * W[(size_t)j * 256 + o];
    out[idx] = s;
}

extern "C" void kernel_launch(void* const* d_in, const int* in_sizes, int n_in,
                              void* d_out, int out_size, void* d_ws, size_t ws_size,
                              hipStream_t stream) {
    (void)in_sizes; (void)n_in; (void)out_size; (void)ws_size;
    const float* x      = (const float*)d_in[0];
    const int*   e0     = (const int*)d_in[1];
    const int*   e1     = (const int*)d_in[2];
    const int*   batch  = (const int*)d_in[3];
    const float* W_k1   = (const float*)d_in[4];
    const float* W_q1   = (const float*)d_in[5];
    const float* W_v1   = (const float*)d_in[6];
    const float* a_rel1 = (const float*)d_in[7];
    const float* m_rel1 = (const float*)d_in[8];
    const float* W_a1   = (const float*)d_in[9];
    const float* W_k23  = (const float*)d_in[10];
    const float* W_q23  = (const float*)d_in[11];
    const float* W_v23  = (const float*)d_in[12];
    const float* a_rel23= (const float*)d_in[13];
    const float* m_rel23= (const float*)d_in[14];
    const float* W_a23  = (const float*)d_in[15];
    const float* W_m1   = (const float*)d_in[16];
    const float* W_m2   = (const float*)d_in[17];
    const float* b_k1   = (const float*)d_in[18];
    const float* b_q1   = (const float*)d_in[19];
    const float* b_v1   = (const float*)d_in[20];
    const float* b_a1   = (const float*)d_in[21];
    const float* b_k23  = (const float*)d_in[22];
    const float* b_q23  = (const float*)d_in[23];
    const float* b_v23  = (const float*)d_in[24];
    const float* b_a23  = (const float*)d_in[25];
    const float* skip23 = (const float*)d_in[26];
    const float* b_m1   = (const float*)d_in[27];
    const float* b_m2   = (const float*)d_in[28];
    const float* p_rel1 = (const float*)d_in[29];
    const float* p_rel23= (const float*)d_in[30];

    // ---- workspace layout ----
    char* w = (char*)d_ws;
    size_t off = 0;
    auto alloc = [&](size_t bytes) -> char* {
        char* p = w + off;
        off = (off + bytes + 255) & ~(size_t)255;
        return p;
    };
    char* regionX = alloc((size_t)NNODES * FIN * 2);         // Xbf for L1; reused as ping buffer HA
    __bf16* Xbf  = (__bf16*)regionX;
    float*  HAf  = (float*)regionX;                          // alias (valid after L1 kqv done)
    __bf16* HAbf = (__bf16*)(regionX + (size_t)NNODES * HD * 4);
    float*  Qf   = (float*)alloc((size_t)NNODES * HD * 4);
    __bf16* Kbf  = (__bf16*)alloc((size_t)NNODES * HD * 2);
    __bf16* Vbf  = (__bf16*)alloc((size_t)NNODES * HD * 2);
    float*  KR0  = (float*)alloc((size_t)NNODES * HD * 4);   // reused as num0
    float*  KR1  = (float*)alloc((size_t)NNODES * HD * 4);   // reused as num1
    float*  VR0  = (float*)alloc((size_t)NNODES * HD * 4);
    float*  VR1  = (float*)alloc((size_t)NNODES * HD * 4);
    float*  logits = (float*)alloc((size_t)2 * NEDGES * NH * 4);
    unsigned* maxb = (unsigned*)alloc((size_t)2 * NNODES * NH * 4);
    float*  sumb = (float*)alloc((size_t)2 * NNODES * NH * 4);
    __bf16* aggBf = (__bf16*)alloc((size_t)NNODES * HD * 2);
    float*  HBf  = (float*)alloc((size_t)NNODES * HD * 4);
    __bf16* HBbf = (__bf16*)alloc((size_t)NNODES * HD * 2);
    float*  ps   = (float*)alloc((size_t)NG * HD * 4);
    float*  cnt  = (float*)alloc((size_t)NG * 4);
    float*  hid  = (float*)alloc((size_t)NG * HD * 4);
    // packed weights
    const size_t P512 = (size_t)16 * 8 * 32 * 16; // elems, 512x128
    const size_t P128 = (size_t)4 * 8 * 32 * 16;  // elems, 128x128
    __bf16* pWk1 = (__bf16*)alloc(P512 * 2);
    __bf16* pWq1 = (__bf16*)alloc(P512 * 2);
    __bf16* pWv1 = (__bf16*)alloc(P512 * 2);
    __bf16* pWa1 = (__bf16*)alloc(P128 * 2);
    __bf16* pA[3][2], *pM[3][2], *pWk[3], *pWq[3], *pWv[3], *pWa[3];
    pWk[0] = pWk1; pWq[0] = pWq1; pWv[0] = pWv1; pWa[0] = pWa1;
    for (int l = 0; l < 3; l++)
        for (int r = 0; r < 2; r++) {
            pA[l][r] = (__bf16*)alloc(P128 * 2);
            pM[l][r] = (__bf16*)alloc(P128 * 2);
        }
    for (int l = 1; l < 3; l++) {
        pWk[l] = (__bf16*)alloc(P128 * 2);
        pWq[l] = (__bf16*)alloc(P128 * 2);
        pWv[l] = (__bf16*)alloc(P128 * 2);
        pWa[l] = (__bf16*)alloc(P128 * 2);
    }

    // ---- pack all weights ----
    dim3 g512(16, 8), g128(4, 8);
    pack_b<<<g512, 32, 0, stream>>>(W_k1, pWk1, 512, 128);
    pack_b<<<g512, 32, 0, stream>>>(W_q1, pWq1, 512, 128);
    pack_b<<<g512, 32, 0, stream>>>(W_v1, pWv1, 512, 128);
    pack_b<<<g128, 32, 0, stream>>>(W_a1, pWa1, 128, 128);
    for (int r = 0; r < 2; r++) {
        pack_rel<<<g128, 32, 0, stream>>>(a_rel1 + (size_t)r * NH * 64 * 64, pA[0][r]);
        pack_rel<<<g128, 32, 0, stream>>>(m_rel1 + (size_t)r * NH * 64 * 64, pM[0][r]);
    }
    for (int l = 0; l < 2; l++) {
        pack_b<<<g128, 32, 0, stream>>>(W_k23 + (size_t)l * HD * HD, pWk[l + 1], 128, 128);
        pack_b<<<g128, 32, 0, stream>>>(W_q23 + (size_t)l * HD * HD, pWq[l + 1], 128, 128);
        pack_b<<<g128, 32, 0, stream>>>(W_v23 + (size_t)l * HD * HD, pWv[l + 1], 128, 128);
        pack_b<<<g128, 32, 0, stream>>>(W_a23 + (size_t)l * HD * HD, pWa[l + 1], 128, 128);
        for (int r = 0; r < 2; r++) {
            pack_rel<<<g128, 32, 0, stream>>>(a_rel23 + (size_t)(l * 2 + r) * NH * 64 * 64, pA[l + 1][r]);
            pack_rel<<<g128, 32, 0, stream>>>(m_rel23 + (size_t)(l * 2 + r) * NH * 64 * 64, pM[l + 1][r]);
        }
    }

    const int Mt = (NNODES + 15) / 16;
    dim3 gg(Mt, HD / 64); // 16x64 wave tiles over N=128 outputs
    const int EB = (2 * NEDGES * NH + 255) / 256;
    const int NB = (int)(((long)NNODES * HD + 255) / 256);

    auto run_layer = [&](const __bf16* Xin, int F, const float* resid,
                         int l, const float* bk, const float* bq, const float* bv,
                         const float* prel, const float* ba,
                         const float* skipPtr, int skipIdx,
                         float* Houtf, __bf16* Houtbf) {
        // K/Q/V projections (WMMA)
        gemm_wmma<<<gg, 32, 0, stream>>>(Xin, F, pWk[l], F, bk, nullptr, Kbf, HD, NNODES, nullptr, 0, nullptr, 0, 0);
        gemm_wmma<<<gg, 32, 0, stream>>>(Xin, F, pWq[l], F, bq, Qf, nullptr, HD, NNODES, nullptr, 0, nullptr, 0, 0);
        gemm_wmma<<<gg, 32, 0, stream>>>(Xin, F, pWv[l], F, bv, nullptr, Vbf, HD, NNODES, nullptr, 0, nullptr, 0, 0);
        // per-relation head transforms (block-diag WMMA)
        gemm_wmma<<<gg, 32, 0, stream>>>(Kbf, HD, pA[l][0], HD, nullptr, KR0, nullptr, HD, NNODES, nullptr, 0, nullptr, 0, 0);
        gemm_wmma<<<gg, 32, 0, stream>>>(Kbf, HD, pA[l][1], HD, nullptr, KR1, nullptr, HD, NNODES, nullptr, 0, nullptr, 0, 0);
        gemm_wmma<<<gg, 32, 0, stream>>>(Vbf, HD, pM[l][0], HD, nullptr, VR0, nullptr, HD, NNODES, nullptr, 0, nullptr, 0, 0);
        gemm_wmma<<<gg, 32, 0, stream>>>(Vbf, HD, pM[l][1], HD, nullptr, VR1, nullptr, HD, NNODES, nullptr, 0, nullptr, 0, 0);
        // segment softmax: max pass
        hipMemsetAsync(maxb, 0, (size_t)2 * NNODES * NH * 4, stream);
        hipMemsetAsync(sumb, 0, (size_t)2 * NNODES * NH * 4, stream);
        edge_logits<<<EB, 256, 0, stream>>>(e0, e1, Qf, KR0, KR1, prel, logits, maxb);
        // numerator accumulation (KR buffers become num buffers)
        hipMemsetAsync(KR0, 0, (size_t)NNODES * HD * 4, stream);
        hipMemsetAsync(KR1, 0, (size_t)NNODES * HD * 4, stream);
        edge_accum<<<EB, 256, 0, stream>>>(e0, e1, VR0, VR1, logits, maxb, sumb, KR0, KR1);
        node_gelu<<<NB, 256, 0, stream>>>(KR0, KR1, sumb, aggBf);
        // output projection + (skip) + relu (WMMA)
        int mode = skipPtr ? 1 : 2;
        gemm_wmma<<<gg, 32, 0, stream>>>(aggBf, HD, pWa[l], HD, ba, Houtf, Houtbf, HD, NNODES,
                                         resid, HD, skipPtr, skipIdx, mode);
    };

    // layer 1 (512 -> 128, no skip)
    f32_to_bf16<<<(int)(((long)NNODES * FIN + 255) / 256), 256, 0, stream>>>(x, Xbf, (long)NNODES * FIN);
    run_layer(Xbf, FIN, nullptr, 0, b_k1, b_q1, b_v1, p_rel1, b_a1, nullptr, 0, HAf, HAbf);
    // layer 2 (128 -> 128, gated skip, gate index 0)
    run_layer(HAbf, HD, HAf, 1, b_k23, b_q23, b_v23, p_rel23, b_a23, skip23, 0, HBf, HBbf);
    // layer 3 (128 -> 128, gated skip, gate index 1)
    run_layer(HBbf, HD, HBf, 2, b_k23 + HD, b_q23 + HD, b_v23 + HD, p_rel23 + 4, b_a23 + HD, skip23, 1, HAf, HAbf);

    // global mean pool + MLP
    hipMemsetAsync(ps, 0, (size_t)NG * HD * 4, stream);
    hipMemsetAsync(cnt, 0, (size_t)NG * 4, stream);
    pool_sum<<<NB, 256, 0, stream>>>(HAf, batch, ps, cnt);
    mlp1<<<(NG * HD + 255) / 256, 256, 0, stream>>>(ps, cnt, W_m1, b_m1, hid);
    mlp2<<<(NG * 256 + 255) / 256, 256, 0, stream>>>(hid, W_m2, b_m2, (float*)d_out);
}
